// LSTMGenerator_81604378624516
// MI455X (gfx1250) — matrix operand order repaired and checked by reading, not compiled, not run on hardware
//
#include <hip/hip_runtime.h>
#include <hip/hip_bf16.h>

// ---------------------------------------------------------------------------
// Types
// ---------------------------------------------------------------------------
typedef __attribute__((ext_vector_type(16))) _Float16 v16h;
typedef __attribute__((ext_vector_type(8)))  _Float16 v8h;
typedef __attribute__((ext_vector_type(8)))  float    v8f;

// Problem constants
constexpr int IN_DIM = 128;
constexpr int HID    = 256;
constexpr int OUT_DIM= 128;
constexpr int BATCH  = 256;
constexpr int SEQ    = 512;
constexpr int GRID_WGS = 64;   // persistent layer kernel workgroups

// ---------------------------------------------------------------------------
// Workspace layout (bytes, all offsets 256-aligned)
// ---------------------------------------------------------------------------
constexpr size_t OFF_BAR  = 0;                              // 2 * u32 barrier
constexpr size_t OFF_XS0  = 256;                            // x f16 [S,B,128]  32MB
constexpr size_t OFF_H0   = OFF_XS0 + (size_t)SEQ*BATCH*IN_DIM*2;   // h0seq f16 [S,B,256] 64MB
constexpr size_t OFF_H1   = OFF_H0  + (size_t)SEQ*BATCH*HID*2;      // h1seq f16 [S,B,256] 64MB
constexpr size_t OFF_WC0  = OFF_H1  + (size_t)SEQ*BATCH*HID*2;      // Wcat0 f16 [1024,384]
constexpr size_t OFF_WC1  = OFF_WC0 + (size_t)4*HID*(IN_DIM+HID)*2; // Wcat1 f16 [1024,512]
constexpr size_t OFF_WOUT = OFF_WC1 + (size_t)4*HID*(HID+HID)*2;    // Wout f16 [128,256]
constexpr size_t OFF_B0   = OFF_WOUT + (size_t)OUT_DIM*HID*2;       // bias0 f32 [1024]
constexpr size_t OFF_B1   = OFF_B0  + 4096;
constexpr size_t OFF_HBUF = OFF_B1  + 4096;                 // h double buffer f16 [2,256*256]

// ---------------------------------------------------------------------------
// Math helpers (robust at +/- inf)
// ---------------------------------------------------------------------------
__device__ inline float fast_sigmoid(float x) {
    return __builtin_amdgcn_rcpf(1.0f + __expf(-x));
}
__device__ inline float fast_tanh(float x) {
    // tanh(x) = 1 - 2/(e^{2x}+1); saturates correctly for |x| large
    return 1.0f - 2.0f * __builtin_amdgcn_rcpf(__expf(2.0f * x) + 1.0f);
}

// ---------------------------------------------------------------------------
// WMMA fragment loaders (layouts per CDNA5 ISA 7.12.2, wave32)
// A tile 16x32 (MxK), f16: lane m=lane&15, half=lane>>4
//   element e -> k = (e%8) + 16*(e/8) + 8*half     (two contiguous 16B loads)
// B tile 32x16 (KxN), f16: lane n=lane&15, khalf=lane>>4
//   element e -> k = (e%8) + 16*(e/8) + 8*khalf    (two contiguous 16B loads)
// Works for both global and LDS pointers (addrspace inferred after inlining).
// ---------------------------------------------------------------------------
__device__ __forceinline__ v16h load_frag_a(const _Float16* __restrict__ base,
                                            int rowStride, int lane) {
    const _Float16* p = base + (size_t)(lane & 15) * rowStride + ((lane >> 4) << 3);
    v8h lo = *(const v8h*)(p);        // k = 8*half + 0..7
    v8h hi = *(const v8h*)(p + 16);   // k = 16 + 8*half + 0..7
    return __builtin_shufflevector(lo, hi, 0,1,2,3,4,5,6,7,8,9,10,11,12,13,14,15);
}

__device__ __forceinline__ v16h load_frag_b(const _Float16* base,
                                            int rowStride, int lane) {
    const _Float16* p = base + (size_t)(lane & 15) * rowStride + ((lane >> 4) << 3);
    v8h lo = *(const v8h*)(p);        // k = 8*half + 0..7
    v8h hi = *(const v8h*)(p + 16);   // k = 16 + 8*half + 0..7
    return __builtin_shufflevector(lo, hi, 0,1,2,3,4,5,6,7,8,9,10,11,12,13,14,15);
}

__device__ __forceinline__ v8f wmma_f16(v16h a, v16h b, v8f c) {
    return __builtin_amdgcn_wmma_f32_16x16x32_f16(
        /*neg_a=*/false, a, /*neg_b=*/false, b,
        /*c_mod=*/(short)0, c, /*reuse_a=*/false, /*reuse_b=*/false);
}

// ---------------------------------------------------------------------------
// Device-wide arrive/spin barrier (persistent grid of GRID_WGS workgroups)
// bar[0] = arrival counter, bar[1] = generation
// ---------------------------------------------------------------------------
__device__ inline void grid_barrier(unsigned* bar) {
    __syncthreads();
    if (threadIdx.x == 0) {
        __threadfence();  // release our h stores to device scope
        unsigned gen = __hip_atomic_load(&bar[1], __ATOMIC_RELAXED, __HIP_MEMORY_SCOPE_AGENT);
        unsigned prev = __hip_atomic_fetch_add(&bar[0], 1u, __ATOMIC_ACQ_REL, __HIP_MEMORY_SCOPE_AGENT);
        if (prev == GRID_WGS - 1) {
            __hip_atomic_store(&bar[0], 0u, __ATOMIC_RELAXED, __HIP_MEMORY_SCOPE_AGENT);
            __hip_atomic_fetch_add(&bar[1], 1u, __ATOMIC_RELEASE, __HIP_MEMORY_SCOPE_AGENT);
        } else {
            while (__hip_atomic_load(&bar[1], __ATOMIC_RELAXED, __HIP_MEMORY_SCOPE_AGENT) == gen) {
                __builtin_amdgcn_s_sleep(2);
            }
        }
        __threadfence();  // acquire other WGs' h stores
    }
    __syncthreads();
}

// ---------------------------------------------------------------------------
// Prepack kernels
// ---------------------------------------------------------------------------
__global__ void cvt_transpose_x(const float* __restrict__ in, _Float16* __restrict__ out) {
    // in: [B,S,128] f32  ->  out: [S,B,128] f16
    size_t idx = (size_t)blockIdx.x * blockDim.x + threadIdx.x;
    if (idx >= (size_t)BATCH * SEQ * IN_DIM) return;
    int i = (int)(idx % IN_DIM);
    int s = (int)((idx / IN_DIM) % SEQ);
    int b = (int)(idx / ((size_t)IN_DIM * SEQ));
    out[((size_t)s * BATCH + b) * IN_DIM + i] = (_Float16)in[idx];
}

__global__ void pack_wcat(const float* __restrict__ Wih, const float* __restrict__ Whh,
                          _Float16* __restrict__ out, int K1) {
    int KT = K1 + HID;
    size_t idx = (size_t)blockIdx.x * blockDim.x + threadIdx.x;
    if (idx >= (size_t)4 * HID * KT) return;
    int k = (int)(idx % KT);
    int n = (int)(idx / KT);
    float v = (k < K1) ? Wih[(size_t)n * K1 + k] : Whh[(size_t)n * HID + (k - K1)];
    out[idx] = (_Float16)v;
}

__global__ void pack_wout(const float* __restrict__ W, _Float16* __restrict__ out) {
    int idx = blockIdx.x * blockDim.x + threadIdx.x;
    if (idx >= OUT_DIM * HID) return;
    out[idx] = (_Float16)W[idx];
}

__global__ void pack_bias(const float* __restrict__ a, const float* __restrict__ b,
                          float* __restrict__ out) {
    int idx = blockIdx.x * blockDim.x + threadIdx.x;
    if (idx >= 4 * HID) return;
    out[idx] = a[idx] + b[idx];
}

__global__ void init_bar(unsigned* bar) {
    if (threadIdx.x < 2) bar[threadIdx.x] = 0u;
}

// ---------------------------------------------------------------------------
// Persistent LSTM layer kernel.
// Grid: 64 WGs x 128 threads (4 waves). 256 waves total; wave w owns the
// 16x16 tile (mb = w>>4 of batch, nb = w&15 of hidden). Cell state c lives
// in VGPRs across all 512 steps. h double-buffered in global (L2-resident).
// Weights are loop-invariant -> staged once per wave into LDS (192/256 KB
// per WG out of the WGP's 320 KB); B fragments then come from ds_load_b128.
// Per step: gates = [x_t | h_prev] @ Wcat^T + bias, all via v_wmma f16.
// ---------------------------------------------------------------------------
template<int K1>
__launch_bounds__(128, 1)
__global__ void lstm_layer(const _Float16* __restrict__ xseq,  // [S,B,K1]
                           const _Float16* __restrict__ Wcat,  // [4H, K1+256]
                           const float*    __restrict__ bias,  // [4H] combined
                           _Float16*       __restrict__ hseq,  // [S,B,256] out
                           _Float16*       __restrict__ hbuf,  // [2, 256*256]
                           unsigned*       __restrict__ bar) {
    constexpr int KTOT = K1 + HID;
    extern __shared__ _Float16 smem[];   // [4 waves][4 gates][16 rows][KTOT]

    const int lane = threadIdx.x & 31;
    const int wv   = threadIdx.x >> 5;
    const int gw   = blockIdx.x * 4 + wv;   // 0..255
    const int mb   = gw >> 4;               // batch tile (0..15)
    const int nb   = gw & 15;               // hidden-col tile (0..15)
    const int hlf  = lane >> 4;
    const int nloc = lane & 15;
    const int col  = nb * 16 + nloc;        // my hidden column

    // ---- stage this wave's weight rows into LDS (one-time) ----
    _Float16* wlds = smem + (size_t)wv * (4 * 16 * KTOT);
    #pragma unroll
    for (int g = 0; g < 4; ++g) {
        const _Float16* src = Wcat + (size_t)(g * HID + nb * 16) * KTOT;
        _Float16*       dst = wlds + g * (16 * KTOT);
        for (int i = lane * 8; i < 16 * KTOT; i += 32 * 8) {
            *(v8h*)(dst + i) = *(const v8h*)(src + i);
        }
    }

    // Zero my tile of hbuf[0] (h_{-1} = 0); every tile covered exactly once.
    #pragma unroll
    for (int r = 0; r < 8; ++r) {
        int m = mb * 16 + r + hlf * 8;
        hbuf[(size_t)m * HID + col] = (_Float16)0.0f;
    }

    // Per-column gate biases (constant over rows)
    float bg[4];
    #pragma unroll
    for (int g = 0; g < 4; ++g) bg[g] = bias[g * HID + col];

    v8f cst = {};          // cell state tile, register-resident for all steps
    grid_barrier(bar);     // h0 zeros (and LDS fill) visible / complete

    for (int t = 0; t < SEQ; ++t) {
        const int cur = t & 1;
        const int nxt = cur ^ 1;

        const _Float16* Ax = xseq + ((size_t)t * BATCH + mb * 16) * K1;
        const _Float16* Ah = hbuf + (size_t)cur * (BATCH * HID) + (size_t)(mb * 16) * HID;

        // Prefetch next step's x tile rows (stream from HBM behind this step)
        if (t + 1 < SEQ) {
            const _Float16* nx = xseq + ((size_t)(t + 1) * BATCH + mb * 16) * K1;
            __builtin_prefetch(nx + (size_t)nloc * K1 + hlf * 64, 0, 1);
        }

        v8f acc[4];
        #pragma unroll
        for (int g = 0; g < 4; ++g)
            #pragma unroll
            for (int r = 0; r < 8; ++r) acc[g][r] = bg[g];

        // x_t contribution (K = K1): A from global, B from LDS
        #pragma unroll
        for (int kc = 0; kc < K1 / 32; ++kc) {
            v16h a = load_frag_a(Ax + kc * 32, K1, lane);
            #pragma unroll
            for (int g = 0; g < 4; ++g) {
                v16h b = load_frag_b(wlds + g * (16 * KTOT) + kc * 32, KTOT, lane);
                acc[g] = wmma_f16(a, b, acc[g]);
            }
        }
        // h_{t-1} contribution (K = 256): A from global (L2), B from LDS
        #pragma unroll
        for (int kc = 0; kc < HID / 32; ++kc) {
            v16h a = load_frag_a(Ah + kc * 32, HID, lane);
            #pragma unroll
            for (int g = 0; g < 4; ++g) {
                v16h b = load_frag_b(wlds + g * (16 * KTOT) + K1 + kc * 32, KTOT, lane);
                acc[g] = wmma_f16(a, b, acc[g]);
            }
        }

        // LSTM cell (elementwise on the 16x16 tile; C/D layout: M=r+8*hlf, N=nloc)
        #pragma unroll
        for (int r = 0; r < 8; ++r) {
            float iv = fast_sigmoid(acc[0][r]);
            float fv = fast_sigmoid(acc[1][r]);
            float gv = fast_tanh   (acc[2][r]);
            float ov = fast_sigmoid(acc[3][r]);
            float c  = fv * cst[r] + iv * gv;
            cst[r] = c;
            float hv = ov * fast_tanh(c);
            _Float16 hh = (_Float16)hv;
            int m = mb * 16 + r + hlf * 8;
            hbuf[(size_t)nxt * (BATCH * HID) + (size_t)m * HID + col] = hh;
            hseq[((size_t)t * BATCH + m) * HID + col] = hh;
        }

        grid_barrier(bar); // publish h_t before anyone reads it at t+1
    }
}

// ---------------------------------------------------------------------------
// Output head: out[b,s,:] = tanh(h1[s,b,:] @ Wout^T + b_out)
// Grid: 8192 blocks (one per 16-row M tile of the [S*B, 256] matrix),
// 8 waves per block = 8 N tiles covering OUT_DIM=128.
// ---------------------------------------------------------------------------
__launch_bounds__(256, 2)
__global__ void head_kernel(const _Float16* __restrict__ h1,    // [S,B,256]
                            const _Float16* __restrict__ Wout,  // [128,256]
                            const float*    __restrict__ bout,  // [128]
                            float*          __restrict__ out) { // [B,S,128]
    const int lane = threadIdx.x & 31;
    const int wv   = threadIdx.x >> 5;     // ntile 0..7
    const int hlf  = lane >> 4;
    const int nloc = lane & 15;
    const int mtile = blockIdx.x;          // 0..8191
    const int col  = wv * 16 + nloc;

    const _Float16* Abase = h1 + (size_t)mtile * 16 * HID;
    const _Float16* Bbase = Wout + (size_t)(wv * 16) * HID;

    float bv = bout[col];
    v8f acc;
    #pragma unroll
    for (int r = 0; r < 8; ++r) acc[r] = bv;

    #pragma unroll
    for (int kc = 0; kc < HID / 32; ++kc) {
        v16h a = load_frag_a(Abase + kc * 32, HID, lane);
        v16h b = load_frag_b(Bbase + kc * 32, HID, lane);
        acc = wmma_f16(a, b, acc);
    }

    #pragma unroll
    for (int r = 0; r < 8; ++r) {
        int m = mtile * 16 + r + hlf * 8;  // row of [S*B]
        int s = m >> 8;                    // m / 256
        int b = m & 255;                   // m % 256
        out[((size_t)b * SEQ + s) * OUT_DIM + col] = fast_tanh(acc[r]);
    }
}

// ---------------------------------------------------------------------------
// Launch
// ---------------------------------------------------------------------------
extern "C" void kernel_launch(void* const* d_in, const int* in_sizes, int n_in,
                              void* d_out, int out_size, void* d_ws, size_t ws_size,
                              hipStream_t stream) {
    const float* x     = (const float*)d_in[0];
    const float* W_ih0 = (const float*)d_in[1];
    const float* W_hh0 = (const float*)d_in[2];
    const float* b_ih0 = (const float*)d_in[3];
    const float* b_hh0 = (const float*)d_in[4];
    const float* W_ih1 = (const float*)d_in[5];
    const float* W_hh1 = (const float*)d_in[6];
    const float* b_ih1 = (const float*)d_in[7];
    const float* b_hh1 = (const float*)d_in[8];
    const float* W_out = (const float*)d_in[9];
    const float* b_out = (const float*)d_in[10];
    float* out = (float*)d_out;

    char* ws = (char*)d_ws;
    unsigned*  bar   = (unsigned*)(ws + OFF_BAR);
    _Float16*  xs0   = (_Float16*)(ws + OFF_XS0);
    _Float16*  h0s   = (_Float16*)(ws + OFF_H0);
    _Float16*  h1s   = (_Float16*)(ws + OFF_H1);
    _Float16*  wc0   = (_Float16*)(ws + OFF_WC0);
    _Float16*  wc1   = (_Float16*)(ws + OFF_WC1);
    _Float16*  wout  = (_Float16*)(ws + OFF_WOUT);
    float*     bias0 = (float*)   (ws + OFF_B0);
    float*     bias1 = (float*)   (ws + OFF_B1);
    _Float16*  hbuf  = (_Float16*)(ws + OFF_HBUF);

    // --- prepack (f32 -> f16, transpose x, fuse W_ih|W_hh, combine biases) ---
    {
        size_t n = (size_t)BATCH * SEQ * IN_DIM;
        cvt_transpose_x<<<(unsigned)((n + 255) / 256), 256, 0, stream>>>(x, xs0);
    }
    {
        size_t n0 = (size_t)4 * HID * (IN_DIM + HID);
        pack_wcat<<<(unsigned)((n0 + 255) / 256), 256, 0, stream>>>(W_ih0, W_hh0, wc0, IN_DIM);
        size_t n1 = (size_t)4 * HID * (HID + HID);
        pack_wcat<<<(unsigned)((n1 + 255) / 256), 256, 0, stream>>>(W_ih1, W_hh1, wc1, HID);
    }
    pack_wout<<<(OUT_DIM * HID + 255) / 256, 256, 0, stream>>>(W_out, wout);
    pack_bias<<<(4 * HID + 255) / 256, 256, 0, stream>>>(b_ih0, b_hh0, bias0);
    pack_bias<<<(4 * HID + 255) / 256, 256, 0, stream>>>(b_ih1, b_hh1, bias1);
    init_bar<<<1, 32, 0, stream>>>(bar);

    // --- layer 0 and layer 1 (persistent, one grid barrier per timestep) ---
    constexpr size_t shmem0 = (size_t)4 * 4 * 16 * (IN_DIM + HID) * sizeof(_Float16); // 192 KB
    constexpr size_t shmem1 = (size_t)4 * 4 * 16 * (HID + HID) * sizeof(_Float16);    // 256 KB
    lstm_layer<IN_DIM><<<GRID_WGS, 128, shmem0, stream>>>(xs0, wc0, bias0, h0s, hbuf, bar);
    lstm_layer<HID>   <<<GRID_WGS, 128, shmem1, stream>>>(h0s, wc1, bias1, h1s, hbuf, bar);

    // --- output head ---
    head_kernel<<<(SEQ * BATCH) / 16, 256, 0, stream>>>(h1s, wout, b_out, out);
}
